// LSTM_36928128811412
// MI455X (gfx1250) — matrix operand order repaired and checked by reading, not compiled, not run on hardware
//
#include <hip/hip_runtime.h>
#include <stdint.h>

// ---------------- problem sizes ----------------
static constexpr int T = 2048, B = 64, E = 256, H = 512, O = 2;
static constexpr int G = 4 * H;        // 2048 gate columns (i,f,g,o)
static constexpr int KD = E + H;       // 768 fused K (x | h)
static constexpr int NBLK = 32;        // persistent workgroups
static constexpr int HS = H / NBLK;    // 16 h-columns per block
static constexpr int THREADS = 512;    // 16 waves (wave32)
static constexpr int WROW = KD + 8;    // padded LDS weight row (bf16): 776, bank-conflict-free
static constexpr int XROW = E + 8;     // padded LDS x row (bf16): 264, 132 dw stride -> +4 banks/lane
static constexpr int CEXROW = 68;      // padded C-exchange row (f32)

// ---------------- workspace layout (bytes) ----------------
static constexpr size_t WS_WBF  = 0;                                  // G*KD bf16  = 3 MB
static constexpr size_t WS_BIAS = WS_WBF  + (size_t)G * KD * 2;       // G f32
static constexpr size_t WS_H0   = WS_BIAS + (size_t)G * 4;            // B*H bf16
static constexpr size_t WS_H1   = WS_H0   + (size_t)B * H * 2;        // B*H bf16
static constexpr size_t WS_HFIN = WS_H1   + (size_t)B * H * 2;        // B*H f32
static constexpr size_t WS_SYNC = WS_HFIN + (size_t)B * H * 4;        // 2 u32

// ---------------- LDS layout (bytes) ----------------
static constexpr size_t SZ_W   = (size_t)4 * HS * WROW * 2;           // 64 x 776 bf16 = 99328
static constexpr size_t SZ_X   = (size_t)B * XROW * 2;                // 64 x 264 bf16 = 33792
static constexpr size_t SZ_CEX = (size_t)B * CEXROW * 4;              // 64 x 68 f32  = 17408
static constexpr size_t SZ_CST = (size_t)B * HS * 4;                  // 64 x 16 f32  =  4096
static constexpr size_t LDS_BYTES = SZ_W + SZ_X + SZ_CEX + SZ_CST;    // 154624 <= 160KB

typedef __attribute__((ext_vector_type(16))) __bf16 v16bf;
typedef __attribute__((ext_vector_type(8)))  float  v8f;

__device__ __forceinline__ uint16_t f2bf(float f) {
  uint32_t u = __float_as_uint(f);
  u += 0x7FFFu + ((u >> 16) & 1u);      // round-to-nearest-even (prep/update only)
  return (uint16_t)(u >> 16);
}
// Pack two f32 -> packed bf16 dword with one v_perm_b32 (round-to-nearest, ties away).
__device__ __forceinline__ uint32_t pk2bf(float a, float b) {
  uint32_t ua = __float_as_uint(a) + 0x8000u;
  uint32_t ub = __float_as_uint(b) + 0x8000u;
  return __builtin_amdgcn_perm(ub, ua, 0x07060302u);  // {ub[3],ub[2],ua[3],ua[2]}
}
__device__ __forceinline__ float sigmoidf_(float x) {
  return 1.0f / (1.0f + __expf(-x));
}

union Frag { uint32_t u[8]; v16bf v; };

__device__ __forceinline__ void ld_frag_lds(Frag& f, const uint16_t* p0, const uint16_t* p1) {
  uint4 a0 = *(const uint4*)p0;
  uint4 a1 = *(const uint4*)p1;
  f.u[0] = a0.x; f.u[1] = a0.y; f.u[2] = a0.z; f.u[3] = a0.w;
  f.u[4] = a1.x; f.u[5] = a1.y; f.u[6] = a1.z; f.u[7] = a1.w;
}

// ---- prep: combined [G x KD] bf16 weight matrix (row g = [W_ih row | W_hh row]) ----
__global__ void k_prep_w(const float* __restrict__ Wih, const float* __restrict__ Whh,
                         uint16_t* __restrict__ Wbf) {
  int idx = blockIdx.x * blockDim.x + threadIdx.x;
  if (idx < G * KD) {
    int g = idx / KD, k = idx % KD;
    float v = (k < E) ? Wih[(size_t)g * E + k] : Whh[(size_t)g * H + (k - E)];
    Wbf[idx] = f2bf(v);
  }
}

// ---- prep: combined bias, zero h0, zero grid-sync counters ----
__global__ void k_prep_misc(const float* __restrict__ bih, const float* __restrict__ bhh,
                            float* __restrict__ bias, uint16_t* __restrict__ h0,
                            uint32_t* __restrict__ sync) {
  int idx = blockIdx.x * blockDim.x + threadIdx.x;
  if (idx < B * H) h0[idx] = 0;
  if (idx < G) bias[idx] = bih[idx] + bhh[idx];
  if (idx < 2) sync[idx] = 0u;
}

// ---- persistent LSTM recurrence: 32 blocks, grid-barrier per timestep ----
__global__ __launch_bounds__(THREADS) void k_lstm(
    const float* __restrict__ x, const uint16_t* __restrict__ Wbf,
    const float* __restrict__ bias, uint16_t* __restrict__ hb0,
    uint16_t* __restrict__ hb1, float* __restrict__ hfin,
    uint32_t* __restrict__ sync) {
  extern __shared__ __align__(16) char smem[];
  uint16_t* Wl  = (uint16_t*)smem;                         // [64][WROW] bf16, row r = q*16+j
  uint16_t* Xl  = (uint16_t*)(smem + SZ_W);                // [64][XROW] bf16, x_t staged
  float*    Cex = (float*)(smem + SZ_W + SZ_X);            // [64][CEXROW] f32 gate exchange
  float*    cst = (float*)(smem + SZ_W + SZ_X + SZ_CEX);   // [64][16] f32 cell state

  const int tid  = threadIdx.x;
  const int blk  = blockIdx.x;
  const int lane = tid & 31;
  const int wv   = tid >> 5;       // 0..15
  const int mt   = wv >> 2;        // M tile 0..3 (batch rows)
  const int q    = wv & 3;         // gate index == N tile (i,f,g,o)
  const int half = lane >> 4;      // K-half selector per A/B layout
  const int lm   = lane & 15;

  // Stage this block's 64 gate rows (4 gates x 16 cols) of weights into LDS.
  {
    const int urow = KD / 2;  // 384 dwords per source row
    for (int i = tid; i < 4 * HS * urow; i += THREADS) {
      int r = i / urow, c = i % urow;
      int g = (r >> 4) * H + blk * HS + (r & 15);
      ((uint32_t*)(Wl + (size_t)r * WROW))[c] =
          ((const uint32_t*)(Wbf + (size_t)g * KD))[c];
    }
  }
  for (int i = tid; i < B * HS; i += THREADS) cst[i] = 0.0f;
  __syncthreads();

  const float bv = bias[q * H + blk * HS + lm];    // per-N bias (C layout: N = lane%16)
  const int   m  = mt * 16 + lm;                   // A-matrix row for this lane

  const uint16_t* arowx = Xl + (size_t)m * XROW;
  const uint16_t* wrow  = Wl + (size_t)(q * 16 + lm) * WROW;

  for (int t = 0; t < T; ++t) {
    const uint16_t* hcur  = (t & 1) ? hb1 : hb0;
    uint16_t*       hnext = (t & 1) ? hb0 : hb1;

    // Prefetch next timestep's x tile (64KB across 512 lanes) -> global_prefetch_b8
    if (t + 1 < T)
      __builtin_prefetch(x + ((size_t)(t + 1) * B) * E + (size_t)tid * 32, 0, 1);

    // ---- cooperative stage of x_t as bf16 into LDS (8 float4 per thread) ----
    {
      const float4* xsrc = (const float4*)(x + (size_t)t * B * E);
      for (int i = tid; i < (B * E) / 4; i += THREADS) {
        int row = i >> 6;            // 64 float4 per 256-wide row
        int c4  = i & 63;
        float4 f = xsrc[i];
        uint2 d;
        d.x = pk2bf(f.x, f.y);
        d.y = pk2bf(f.z, f.w);
        *(uint2*)(Xl + (size_t)row * XROW + c4 * 4) = d;
      }
    }
    __syncthreads();

    // Two independent accumulation chains (even/odd K blocks) -> 2x WMMA ILP,
    // so fragment loads of one chain overlap the other chain's WMMA.
    v8f acc0, acc1;
    #pragma unroll
    for (int r = 0; r < 8; ++r) { acc0[r] = bv; acc1[r] = 0.0f; }

    const uint16_t* hrow = hcur + (size_t)m * H;

    // ---- K part 1: x contribution, A from LDS (fully unrolled, dual chain) ----
    #pragma unroll
    for (int kb = 0; kb < E; kb += 64) {
      Frag a0, b0, a1, b1;
      ld_frag_lds(a0, arowx + kb + half * 8,       arowx + kb + 16 + half * 8);
      ld_frag_lds(b0, wrow + kb + half * 8,        wrow + kb + 16 + half * 8);
      ld_frag_lds(a1, arowx + kb + 32 + half * 8,  arowx + kb + 48 + half * 8);
      ld_frag_lds(b1, wrow + kb + 32 + half * 8,   wrow + kb + 48 + half * 8);
      acc0 = __builtin_amdgcn_wmma_f32_16x16x32_bf16(
          false, a0.v, false, b0.v, (short)0, acc0, false, false);
      acc1 = __builtin_amdgcn_wmma_f32_16x16x32_bf16(
          false, a1.v, false, b1.v, (short)0, acc1, false, false);
    }
    // ---- K part 2: h contribution, A from global bf16 h (fully unrolled, dual chain) ----
    #pragma unroll
    for (int kb = 0; kb < H; kb += 64) {
      Frag a0, b0, a1, b1;
      ld_frag_lds(a0, hrow + kb + half * 8,        hrow + kb + 16 + half * 8);
      ld_frag_lds(b0, wrow + E + kb + half * 8,    wrow + E + kb + 16 + half * 8);
      ld_frag_lds(a1, hrow + kb + 32 + half * 8,   hrow + kb + 48 + half * 8);
      ld_frag_lds(b1, wrow + E + kb + 32 + half * 8, wrow + E + kb + 48 + half * 8);
      acc0 = __builtin_amdgcn_wmma_f32_16x16x32_bf16(
          false, a0.v, false, b0.v, (short)0, acc0, false, false);
      acc1 = __builtin_amdgcn_wmma_f32_16x16x32_bf16(
          false, a1.v, false, b1.v, (short)0, acc1, false, false);
    }

    // Exchange gate tiles through LDS (i,f,g,o live in different waves).
    {
      int mrow = mt * 16 + half * 8;
      int col  = q * 16 + lm;
      #pragma unroll
      for (int r = 0; r < 8; ++r)
        Cex[(size_t)(mrow + r) * CEXROW + col] = acc0[r] + acc1[r];
    }
    __syncthreads();

    // Nonlinear cell/hidden update for this block's 64x16 slice (2 elems/thread).
    for (int e = tid; e < B * HS; e += THREADS) {
      int b = e >> 4, j = e & 15;
      float gi = Cex[(size_t)b * CEXROW + j];
      float gf = Cex[(size_t)b * CEXROW + 16 + j];
      float gg = Cex[(size_t)b * CEXROW + 32 + j];
      float go = Cex[(size_t)b * CEXROW + 48 + j];
      float c  = cst[e];
      c = sigmoidf_(gf) * c + sigmoidf_(gi) * tanhf(gg);
      cst[e] = c;
      float h = sigmoidf_(go) * tanhf(c);
      hnext[(size_t)b * H + blk * HS + j] = f2bf(h);
      if (t == T - 1) hfin[(size_t)b * H + blk * HS + j] = h;
    }

    // Grid-wide barrier so every block sees the full new h before step t+1.
    __syncthreads();
    if (tid == 0) {
      __threadfence();
      uint32_t prev = __hip_atomic_fetch_add(&sync[0], 1u, __ATOMIC_ACQ_REL,
                                             __HIP_MEMORY_SCOPE_AGENT);
      if (prev == NBLK - 1) {
        __hip_atomic_store(&sync[0], 0u, __ATOMIC_RELAXED, __HIP_MEMORY_SCOPE_AGENT);
        __hip_atomic_fetch_add(&sync[1], 1u, __ATOMIC_RELEASE, __HIP_MEMORY_SCOPE_AGENT);
      } else {
        while (__hip_atomic_load(&sync[1], __ATOMIC_ACQUIRE,
                                 __HIP_MEMORY_SCOPE_AGENT) < (uint32_t)(t + 1))
          __builtin_amdgcn_s_sleep(8);
      }
    }
    __syncthreads();
  }
}

// ---- tiny final projection: pred = sigmoid(h @ W_lin^T + b_lin), 64x2 outputs ----
__global__ void k_final(const float* __restrict__ hfin, const float* __restrict__ Wlin,
                        const float* __restrict__ blin, float* __restrict__ out) {
  int tid = threadIdx.x;
  if (tid < B * O) {
    int b = tid >> 1, o = tid & 1;
    const float* hr = hfin + (size_t)b * H;
    const float* wr = Wlin + (size_t)o * H;
    float s = blin[o];
    for (int k = 0; k < H; ++k) s += hr[k] * wr[k];
    out[b * O + o] = sigmoidf_(s);
  }
}

extern "C" void kernel_launch(void* const* d_in, const int* in_sizes, int n_in,
                              void* d_out, int out_size, void* d_ws, size_t ws_size,
                              hipStream_t stream) {
  (void)in_sizes; (void)n_in; (void)out_size; (void)ws_size;
  const float* data = (const float*)d_in[0];
  const float* Wih  = (const float*)d_in[1];
  const float* Whh  = (const float*)d_in[2];
  const float* bih  = (const float*)d_in[3];
  const float* bhh  = (const float*)d_in[4];
  const float* Wlin = (const float*)d_in[5];
  const float* blin = (const float*)d_in[6];

  char* ws = (char*)d_ws;
  uint16_t* Wbf  = (uint16_t*)(ws + WS_WBF);
  float*    bias = (float*)(ws + WS_BIAS);
  uint16_t* h0   = (uint16_t*)(ws + WS_H0);
  uint16_t* h1   = (uint16_t*)(ws + WS_H1);
  float*    hfin = (float*)(ws + WS_HFIN);
  uint32_t* sync = (uint32_t*)(ws + WS_SYNC);

  k_prep_w<<<(G * KD + 255) / 256, 256, 0, stream>>>(Wih, Whh, Wbf);
  k_prep_misc<<<(B * H + 255) / 256, 256, 0, stream>>>(bih, bhh, bias, h0, sync);
  k_lstm<<<NBLK, THREADS, LDS_BYTES, stream>>>(data, Wbf, bias, h0, h1, hfin, sync);
  k_final<<<1, 128, 0, stream>>>(hfin, Wlin, blin, (float*)d_out);
}
